// TGNNCell_77197742178345
// MI455X (gfx1250) — compile-verified, matrix operand order checked by reading, not run on hardware
//
#include <hip/hip_runtime.h>
#include <cmath>

typedef __attribute__((ext_vector_type(16))) __bf16 v16bf;
typedef __attribute__((ext_vector_type(8)))  __bf16 v8bf;
typedef __attribute__((ext_vector_type(8)))  float  v8f;

constexpr int N_NODES = 100000;
constexpr int N_EDGES = 1000000;
constexpr int IN_CH   = 64;
constexpr int HID     = 64;
constexpr int GIN     = IN_CH + HID;   // 128
constexpr int G3H     = 3 * HID;       // 192

constexpr size_t AL256(size_t v) { return (v + 255) & ~size_t(255); }
constexpr size_t OFF_DEG  = 0;
constexpr size_t OFF_DINV = AL256(OFF_DEG  + size_t(N_NODES) * 4);
constexpr size_t OFF_XW   = AL256(OFF_DINV + size_t(N_NODES) * 4);
constexpr size_t OFF_AGG  = AL256(OFF_XW   + size_t(N_NODES) * HID * 4);
constexpr size_t OFF_G    = AL256(OFF_AGG  + size_t(N_NODES) * HID * 4);
constexpr size_t OFF_WGT  = AL256(OFF_G    + size_t(N_NODES) * HID * 4);  // W_gcn^T bf16
constexpr size_t OFF_WIH  = AL256(OFF_WGT  + size_t(IN_CH) * HID * 2);
constexpr size_t OFF_WHH  = AL256(OFF_WIH  + size_t(G3H) * GIN * 2);

__device__ __forceinline__ float sigmoidf_fast(float v) {
    return 1.0f / (1.0f + __expf(-v));
}

// ---- WMMA fragment helpers (ISA 7.12.2 layouts, wave32) -------------------
// Fragment half-pattern: elems 0..7 -> K = kb..kb+7, elems 8..15 -> K = kb+16..kb+23,
// where kb = k0 + (lane>=16 ? 8 : 0). Both halves are contiguous 16-byte runs.
__device__ __forceinline__ v16bf frag_from_contig(const __bf16* p) {
    v8bf lo = *(const v8bf*)p;
    v8bf hi = *(const v8bf*)(p + 16);
    return __builtin_shufflevector(lo, hi, 0, 1, 2, 3, 4, 5, 6, 7,
                                   8, 9, 10, 11, 12, 13, 14, 15);
}

// A-matrix 16x32 bf16 fragment built from an fp32 row-major source.
__device__ __forceinline__ v16bf load_a_f32(const float* __restrict__ src,
                                            int ld, int row0, int k0, int lane) {
    int m  = row0 + (lane & 15);
    int kb = k0 + ((lane >> 4) << 3);
    const float* p = src + (size_t)m * ld;
    v16bf a;
#pragma unroll
    for (int i = 0; i < 8; ++i) a[i] = (__bf16)p[kb + i];
#pragma unroll
    for (int i = 0; i < 8; ++i) a[8 + i] = (__bf16)p[kb + 16 + i];
    return a;
}

__device__ __forceinline__ v8f wmma_bf16(v16bf a, v16bf b, v8f c) {
    return __builtin_amdgcn_wmma_f32_16x16x32_bf16(
        /*neg_a=*/false, a, /*neg_b=*/false, b,
        /*c_mod=*/(short)0, c, /*reuse_a=*/false, /*reuse_b=*/false);
}

// ---- K1: init deg (self loop = 1), zero agg, convert weights to bf16 ------
// W_gcn is [in][out]; store bf16 copy TRANSPOSED ([out][in]) so B-fragment
// halves are contiguous. W_ih/W_hh (torch [out][in]) are already contiguous.
__global__ void k_init(float* __restrict__ deg, float* __restrict__ agg,
                       const float* __restrict__ Wg_f, const float* __restrict__ Wih_f,
                       const float* __restrict__ Whh_f,
                       __bf16* __restrict__ Wgt, __bf16* __restrict__ Wih,
                       __bf16* __restrict__ Whh) {
    int i = blockIdx.x * blockDim.x + threadIdx.x;
    if (i < N_NODES * HID) agg[i] = 0.0f;
    if (i < N_NODES)       deg[i] = 1.0f;
    if (i < IN_CH * HID)   Wgt[i] = (__bf16)Wg_f[(i & 63) * HID + (i >> 6)];
    if (i < G3H * GIN)     Wih[i] = (__bf16)Wih_f[i];
    if (i < G3H * HID)     Whh[i] = (__bf16)Whh_f[i];
}

// ---- K2: degree accumulation over dst -------------------------------------
__global__ void k_deg(const long long* __restrict__ ei,
                      const float* __restrict__ ew, float* __restrict__ deg) {
    int e = blockIdx.x * blockDim.x + threadIdx.x;
    if (e >= N_EDGES) return;
    long long d = ei[N_EDGES + e];
    atomicAdd(&deg[d], ew[e]);
}

// ---- K3: dinv = rsqrt(deg) ------------------------------------------------
__global__ void k_dinv(const float* __restrict__ deg, float* __restrict__ dinv) {
    int i = blockIdx.x * blockDim.x + threadIdx.x;
    if (i >= N_NODES) return;
    dinv[i] = __frsqrt_rn(deg[i]);
}

// ---- K4: xw = x @ W_gcn via bf16 WMMA; 1 wave per 16-row tile -------------
__global__ void __launch_bounds__(32) k_gcn_gemm(const float* __restrict__ x,
                                                 const __bf16* __restrict__ Wgt,
                                                 float* __restrict__ xw) {
    int lane = threadIdx.x;
    int row0 = blockIdx.x * 16;
    v16bf a0 = load_a_f32(x, IN_CH, row0, 0,  lane);
    v16bf a1 = load_a_f32(x, IN_CH, row0, 32, lane);
    int nrow = lane & 15;                 // B column carried by this lane
    int kb8  = (lane >> 4) << 3;
    v8f c[4] = {};
#pragma unroll
    for (int nt = 0; nt < 4; ++nt) {
        const __bf16* bp = Wgt + (size_t)(nt * 16 + nrow) * IN_CH + kb8;
        c[nt] = wmma_bf16(a0, frag_from_contig(bp),      c[nt]);
        c[nt] = wmma_bf16(a1, frag_from_contig(bp + 32), c[nt]);
    }
    int col   = lane & 15;
    int rbase = (lane >> 4) << 3;
#pragma unroll
    for (int nt = 0; nt < 4; ++nt)
#pragma unroll
        for (int r = 0; r < 8; ++r)
            xw[(size_t)(row0 + rbase + r) * HID + nt * 16 + col] = c[nt][r];
}

// ---- K5: edge scatter: agg[dst] += coef * xw[src] (16 threads / edge) -----
__global__ void k_scatter(const long long* __restrict__ ei,
                          const float* __restrict__ ew,
                          const float* __restrict__ dinv,
                          const float* __restrict__ xw,
                          float* __restrict__ agg) {
    int t = blockIdx.x * blockDim.x + threadIdx.x;
    int e  = t >> 4;
    int j4 = (t & 15) << 2;
    if (e >= N_EDGES) return;
    long long s = ei[e];
    long long d = ei[N_EDGES + e];
    float coef = ew[e] * dinv[s] * dinv[d];
    const float4 xv = *(const float4*)(xw + (size_t)s * HID + j4);
    float* ap = agg + (size_t)d * HID + j4;
    atomicAdd(ap + 0, coef * xv.x);
    atomicAdd(ap + 1, coef * xv.y);
    atomicAdd(ap + 2, coef * xv.z);
    atomicAdd(ap + 3, coef * xv.w);
}

// ---- K6: g = sigmoid(agg + xw/deg + b_gcn) --------------------------------
__global__ void k_gcn_fin(const float* __restrict__ agg, const float* __restrict__ xw,
                          const float* __restrict__ dinv, const float* __restrict__ bg,
                          float* __restrict__ g) {
    int i = blockIdx.x * blockDim.x + threadIdx.x;
    if (i >= N_NODES * HID) return;
    int node = i >> 6;
    float di = dinv[node];
    g[i] = sigmoidf_fast(agg[i] + xw[i] * di * di + bg[i & 63]);
}

// ---- K7: fused GRU cell; 4 waves/block share one 16-row tile --------------
// All waves consume the SAME 16 rows of inp=[x|g] and h, so stage them once
// in LDS as bf16 (one global load + one convert per element instead of four).
__global__ void __launch_bounds__(128) k_gru(const float* __restrict__ x,
                                             const float* __restrict__ g,
                                             const float* __restrict__ h,
                                             const __bf16* __restrict__ Wih,
                                             const __bf16* __restrict__ Whh,
                                             const float* __restrict__ bih,
                                             const float* __restrict__ bhh,
                                             float* __restrict__ out) {
    __shared__ __align__(16) __bf16 s_inp[16 * GIN];   // 4 KB
    __shared__ __align__(16) __bf16 s_h[16 * HID];     // 2 KB

    int t    = threadIdx.x;
    int row0 = blockIdx.x * 16;

    // Stage inp = [x | g]: 2048 bf16, float4 granularity (64-col boundary is
    // float4-aligned so each float4 comes wholly from x or wholly from g).
#pragma unroll
    for (int base = 0; base < 16 * GIN; base += 128 * 4) {
        int idx = base + t * 4;
        int r = idx >> 7;            // /128
        int c = idx & 127;
        const float* src = (c < IN_CH)
            ? (x + (size_t)(row0 + r) * IN_CH + c)
            : (g + (size_t)(row0 + r) * HID + (c - IN_CH));
        float4 v = *(const float4*)src;
        s_inp[idx + 0] = (__bf16)v.x;
        s_inp[idx + 1] = (__bf16)v.y;
        s_inp[idx + 2] = (__bf16)v.z;
        s_inp[idx + 3] = (__bf16)v.w;
    }
    // Stage h tile: 1024 bf16.
#pragma unroll
    for (int base = 0; base < 16 * HID; base += 128 * 4) {
        int idx = base + t * 4;
        int r = idx >> 6;            // /64
        int c = idx & 63;
        float4 v = *(const float4*)(h + (size_t)(row0 + r) * HID + c);
        s_h[idx + 0] = (__bf16)v.x;
        s_h[idx + 1] = (__bf16)v.y;
        s_h[idx + 2] = (__bf16)v.z;
        s_h[idx + 3] = (__bf16)v.w;
    }
    __syncthreads();

    int wave = t >> 5;
    int lane = t & 31;
    int n0   = wave * 16;
    int m    = lane & 15;
    int kb8  = (lane >> 4) << 3;

    v8f gi[3] = {};
    v8f gh[3] = {};

    // gi = inp @ Wih^T for columns n0+{0,64,128}; K = 128 (4 chunks of 32)
#pragma unroll
    for (int kc = 0; kc < 4; ++kc) {
        int k0 = kc * 32;
        v16bf a = frag_from_contig(&s_inp[m * GIN + k0 + kb8]);
#pragma unroll
        for (int gate = 0; gate < 3; ++gate) {
            const __bf16* bp = Wih + (size_t)(n0 + gate * HID + m) * GIN + k0 + kb8;
            gi[gate] = wmma_bf16(a, frag_from_contig(bp), gi[gate]);
        }
    }
    // gh = h @ Whh^T; K = 64 (2 chunks of 32)
#pragma unroll
    for (int kc = 0; kc < 2; ++kc) {
        int k0 = kc * 32;
        v16bf a = frag_from_contig(&s_h[m * HID + k0 + kb8]);
#pragma unroll
        for (int gate = 0; gate < 3; ++gate) {
            const __bf16* bp = Whh + (size_t)(n0 + gate * HID + m) * HID + k0 + kb8;
            gh[gate] = wmma_bf16(a, frag_from_contig(bp), gh[gate]);
        }
    }

    int col = n0 + (lane & 15);
    float bir = bih[col],       bhr = bhh[col];
    float biz = bih[col + 64],  bhz = bhh[col + 64];
    float bin = bih[col + 128], bhn = bhh[col + 128];
    int rbase = (lane >> 4) << 3;
#pragma unroll
    for (int r = 0; r < 8; ++r) {
        int row = row0 + rbase + r;
        float rg = sigmoidf_fast((gi[0][r] + bir) + (gh[0][r] + bhr));
        float z  = sigmoidf_fast((gi[1][r] + biz) + (gh[1][r] + bhz));
        float nn = tanhf((gi[2][r] + bin) + rg * (gh[2][r] + bhn));
        float hv = h[(size_t)row * HID + col];   // fp32 h for the blend
        out[(size_t)row * HID + col] = (1.0f - z) * nn + z * hv;
    }
}

extern "C" void kernel_launch(void* const* d_in, const int* in_sizes, int n_in,
                              void* d_out, int out_size, void* d_ws, size_t ws_size,
                              hipStream_t stream) {
    const float*     x     = (const float*)d_in[0];
    const long long* ei    = (const long long*)d_in[1];
    const float*     ew    = (const float*)d_in[2];
    const float*     h     = (const float*)d_in[3];
    const float*     Wg_f  = (const float*)d_in[4];
    const float*     bg    = (const float*)d_in[5];
    const float*     Wih_f = (const float*)d_in[6];
    const float*     Whh_f = (const float*)d_in[7];
    const float*     bih   = (const float*)d_in[8];
    const float*     bhh   = (const float*)d_in[9];
    float* out = (float*)d_out;

    char* ws = (char*)d_ws;
    float*  deg  = (float*)(ws + OFF_DEG);
    float*  dinv = (float*)(ws + OFF_DINV);
    float*  xw   = (float*)(ws + OFF_XW);
    float*  agg  = (float*)(ws + OFF_AGG);
    float*  g    = (float*)(ws + OFF_G);
    __bf16* Wgt  = (__bf16*)(ws + OFF_WGT);
    __bf16* Wih  = (__bf16*)(ws + OFF_WIH);
    __bf16* Whh  = (__bf16*)(ws + OFF_WHH);

    const int NH = N_NODES * HID;

    k_init<<<(NH + 255) / 256, 256, 0, stream>>>(deg, agg, Wg_f, Wih_f, Whh_f,
                                                 Wgt, Wih, Whh);
    k_deg<<<(N_EDGES + 255) / 256, 256, 0, stream>>>(ei, ew, deg);
    k_dinv<<<(N_NODES + 255) / 256, 256, 0, stream>>>(deg, dinv);
    k_gcn_gemm<<<N_NODES / 16, 32, 0, stream>>>(x, Wgt, xw);
    k_scatter<<<(N_EDGES * 16) / 256, 256, 0, stream>>>(ei, ew, dinv, xw, agg);
    k_gcn_fin<<<(NH + 255) / 256, 256, 0, stream>>>(agg, xw, dinv, bg, g);
    k_gru<<<N_NODES / 16, 128, 0, stream>>>(x, g, h, Wih, Whh, bih, bhh, out);
}
